// MLAAttention_85504208929564
// MI455X (gfx1250) — matrix operand order repaired and checked by reading, not compiled, not run on hardware
//
#include <hip/hip_runtime.h>
#include <hip/hip_bf16.h>

// ---------------- problem constants ----------------
#define HIDDEN   2048
#define NH       16
#define HD       128
#define SEQ      2048
#define BATCH    2
#define MTOT     (BATCH * SEQ)          // 4096 rows of the "token" matrices

typedef _Float16 v4h  __attribute__((ext_vector_type(4)));
typedef _Float16 v8h  __attribute__((ext_vector_type(8)));
typedef _Float16 v16h __attribute__((ext_vector_type(16)));
typedef float    v8f  __attribute__((ext_vector_type(8)));

__device__ __forceinline__ v16h cat8(v8h lo, v8h hi) {
  return __builtin_shufflevector(lo, hi, 0,1,2,3,4,5,6,7,8,9,10,11,12,13,14,15);
}

// A-fragment (16x32 f16) per CDNA5 ISA layout.
// lanes 0-15 : row=lane,    halves 0-7 -> K 0-7,  halves 8-15 -> K 16-23
// lanes 16-31: row=lane-16, halves 0-7 -> K 8-15, halves 8-15 -> K 24-31
__device__ __forceinline__ v16h load_frag_a(const _Float16* base, int ld, int lane) {
  int m = lane & 15, g = lane >> 4;
  const _Float16* p = base + m * ld + g * 8;
  v8h lo = *(const v8h*)(p);
  v8h hi = *(const v8h*)(p + 16);
  return cat8(lo, hi);
}

// B-fragment (32x16 f16) gathered from a tile stored [n][k] (k contiguous).
// lanes 0-15: col=lane, K 0-7,16-23 ; lanes 16-31: col=lane-16, K 8-15,24-31
__device__ __forceinline__ v16h load_frag_b(const _Float16* base, int ld, int lane) {
  int n = lane & 15, g = lane >> 4;
  const _Float16* p = base + n * ld + g * 8;
  v8h lo = *(const v8h*)(p);
  v8h hi = *(const v8h*)(p + 16);
  return cat8(lo, hi);
}

__device__ __forceinline__ v8f wmma_f16(v16h a, v16h b, v8f c) {
  return __builtin_amdgcn_wmma_f32_16x16x32_f16(false, a, false, b, (short)0, c,
                                                false, false);
}

// ---- CDNA5 async global->LDS copy (ASYNCcnt path, no VGPR round-trip) ------
// VDST VGPR carries the LDS byte offset (= low 32 bits of the flat address,
// per the LDS aperture rules); VADDR is the 64-bit global address (GV mode).
__device__ __forceinline__ void async_ld_b128(void* lds, const void* gptr) {
  unsigned int loff = (unsigned int)(unsigned long long)lds;
  asm volatile("global_load_async_to_lds_b128 %0, %1, off"
               :: "v"(loff), "v"(gptr) : "memory");
}
#define WAIT_ASYNC(n) asm volatile("s_wait_asynccnt %0" :: "i"(n) : "memory")

// ---------------- kernel 1a: fp32 -> fp16 cast (plain) ----------------
__global__ void cast_f32_to_f16(const float* __restrict__ in,
                                _Float16* __restrict__ out, int n) {
  int i = (blockIdx.x * blockDim.x + threadIdx.x) * 4;
  if (i + 3 < n) {
    float4 v = *(const float4*)(in + i);
    v4h o = { (_Float16)v.x, (_Float16)v.y, (_Float16)v.z, (_Float16)v.w };
    *(v4h*)(out + i) = o;
  }
}

// ---------------- kernel 1b: fp32 [K][N] -> fp16 [N][K] transpose-cast -------
__global__ __launch_bounds__(256) void cast_transpose_f32_to_f16(
    const float* __restrict__ in,     // [2048][2048] (k-major)
    _Float16* __restrict__ out) {     // [2048][2048] (n-major)
  __shared__ alignas(16) _Float16 T[64][72];
  const int tid = threadIdx.x;
  const int n0 = blockIdx.x * 64, k0 = blockIdx.y * 64;
  {
    int kr = tid >> 2, nc = (tid & 3) * 16;
    const float* src = in + (size_t)(k0 + kr) * HIDDEN + n0 + nc;
    alignas(16) _Float16 tmp[16];
    #pragma unroll
    for (int i = 0; i < 16; i += 4) {
      float4 v = *(const float4*)(src + i);
      tmp[i + 0] = (_Float16)v.x; tmp[i + 1] = (_Float16)v.y;
      tmp[i + 2] = (_Float16)v.z; tmp[i + 3] = (_Float16)v.w;
    }
    *(v8h*)&T[kr][nc]     = *(v8h*)&tmp[0];
    *(v8h*)&T[kr][nc + 8] = *(v8h*)&tmp[8];
  }
  __syncthreads();
  {
    int nr = tid >> 2, kc = (tid & 3) * 16;
    alignas(16) _Float16 tmp[16];
    #pragma unroll
    for (int i = 0; i < 16; ++i) tmp[i] = T[kc + i][nr];
    _Float16* dst = out + (size_t)(n0 + nr) * HIDDEN + k0 + kc;
    *(v8h*)dst       = *(v8h*)&tmp[0];
    *(v8h*)(dst + 8) = *(v8h*)&tmp[8];
  }
}

// ---------------- kernel 2: fused QKV GEMM + RMSNorm + scale + RoPE ----------
// 64x128 tile (one full head in N); double-buffered async LDS staging.
// mode 1: RoPE, out [B,NH,S,HD].  mode 0: no RoPE, out transposed [B,NH,HD,S].
struct QkvStage { _Float16 As[2][64][40]; _Float16 Bt[2][128][40]; };
struct QkvEpi   { float Cs[64][132]; float red[64][2]; };
union  QkvSh    { QkvStage s; QkvEpi e; };

__global__ __launch_bounds__(256) void qkv_gemm_kernel(
    const _Float16* __restrict__ A,       // [4096, 2048] tokens
    const _Float16* __restrict__ Wt,      // [2048, 2048] pre-transposed Wt[n][k]
    const float*    __restrict__ norm_w,  // [128]
    const float*    __restrict__ scale,   // [1]
    _Float16*       __restrict__ out,
    int mode) {
  __shared__ alignas(16) QkvSh sh;

  const int tid  = threadIdx.x;
  const int lane = tid & 31, wid = tid >> 5;
  const int wm = wid & 3, wn = wid >> 2;
  const int m0 = blockIdx.x * 64;
  const int head = blockIdx.y;
  const int n0 = head * HD;

  v8f acc[4];
  #pragma unroll
  for (int j = 0; j < 4; ++j) { v8f z = {}; acc[j] = z; }

  const int ar = tid >> 2, ac = (tid & 3) * 8;       // A slot: 1 x b128
  const int bn = tid & 127, bkh = tid >> 7;          // B slot: 2 x b128

  auto stage = [&](int buf, int kt) {
    async_ld_b128(&sh.s.As[buf][ar][ac],
                  &A[(size_t)(m0 + ar) * HIDDEN + kt + ac]);
    const _Float16* src = &Wt[(size_t)(n0 + bn) * HIDDEN + kt + bkh * 16];
    async_ld_b128(&sh.s.Bt[buf][bn][bkh * 16],     src);
    async_ld_b128(&sh.s.Bt[buf][bn][bkh * 16 + 8], src + 8);
  };

  stage(0, 0);
  for (int kt = 0; kt < HIDDEN; kt += 32) {
    int buf = (kt >> 5) & 1;
    if (kt + 32 < HIDDEN) { stage(buf ^ 1, kt + 32); WAIT_ASYNC(3); }
    else                  { WAIT_ASYNC(0); }
    __syncthreads();

    v16h a = load_frag_a(&sh.s.As[buf][wm * 16][0], 40, lane);
    #pragma unroll
    for (int j = 0; j < 4; ++j) {
      v16h b = load_frag_b(&sh.s.Bt[buf][wn * 64 + j * 16][0], 40, lane);
      acc[j] = wmma_f16(a, b, acc[j]);
    }
    __syncthreads();
  }

  // spill accumulators (C layout: row = (lane>>4)*8 + r, col = lane&15)
  #pragma unroll
  for (int j = 0; j < 4; ++j)
    #pragma unroll
    for (int r = 0; r < 8; ++r)
      sh.e.Cs[wm * 16 + (lane >> 4) * 8 + r][wn * 64 + j * 16 + (lane & 15)] =
          acc[j][r];
  __syncthreads();

  const float sc = scale[0];
  if (tid < 128) {
    int row = tid >> 1, half = tid & 1;
    float ss = 0.f;
    for (int i = 0; i < 64; ++i) { float x = sh.e.Cs[row][half * 64 + i]; ss += x * x; }
    sh.e.red[row][half] = ss;
  }
  __syncthreads();
  if (tid < 128) {
    int row = tid >> 1, half = tid & 1;
    float inv = rsqrtf((sh.e.red[row][0] + sh.e.red[row][1]) * (1.f / (float)HD) + 1e-6f);
    for (int i = 0; i < 64; ++i) {
      int d = half * 64 + i;
      sh.e.Cs[row][d] = norm_w[d] * (sh.e.Cs[row][d] * inv) * sc;
    }
  }
  __syncthreads();
  if (tid < 128) {
    int row = tid >> 1, half = tid & 1;
    int m = m0 + row, b = m >> 11, s = m & (SEQ - 1);
    if (mode == 1) {
      size_t obase = (((size_t)b * NH + head) * SEQ + s) * HD;
      alignas(16) _Float16 tmpo[64];
      for (int i = 0; i < 64; ++i) {
        int d = half * 64 + i;
        float v = sh.e.Cs[row][d];
        int dr = d & 63;
        float theta = (float)s * __powf(10000.f, -(float)dr * (1.f / 64.f));
        float cth = __cosf(theta), sth = __sinf(theta);
        float other = sh.e.Cs[row][(d + 64) & 127];
        v = (d < 64) ? (v * cth - other * sth) : (v * cth + other * sth);
        tmpo[i] = (_Float16)v;
      }
      #pragma unroll
      for (int i = 0; i < 64; i += 8)
        *(v8h*)&out[obase + half * 64 + i] = *(v8h*)&tmpo[i];
    } else {
      // V path: store transposed [B,NH,HD,S]
      size_t obase = (((size_t)b * NH + head) * HD) * SEQ + s;
      for (int i = 0; i < 64; ++i) {
        int d = half * 64 + i;
        out[obase + (size_t)d * SEQ] = (_Float16)sh.e.Cs[row][d];
      }
    }
  }
}

// ---------------- kernel 3: flash attention per (b,h, q-tile of 64) ---------
// Q fragments in registers; K/V double-buffered with async staging.
__global__ __launch_bounds__(256) void attn_kernel(
    const _Float16* __restrict__ Q,   // [B,NH,S,HD]
    const _Float16* __restrict__ K,   // [B,NH,S,HD]
    const _Float16* __restrict__ V,   // [B,NH,HD,S]  (pre-transposed)
    _Float16*       __restrict__ O) { // [B,S,NH,HD]
  __shared__ alignas(16) _Float16 Ks[2][32][136];
  __shared__ alignas(16) _Float16 Vt[2][128][40];     // [d][kv_local]
  __shared__ alignas(16) float    Ss[64][36];
  __shared__ alignas(16) _Float16 Ps[64][40];
  __shared__ float m_l[64], l_l[64], al_l[64];

  const int tid = threadIdx.x, lane = tid & 31, wid = tid >> 5;
  const int bh = blockIdx.y;
  const int q0 = blockIdx.x * 64;
  const size_t base  = (size_t)bh * SEQ * HD;   // Q,K
  const size_t baseV = (size_t)bh * HD * SEQ;   // V (transposed)

  const int om = wid & 3, on = (wid >> 2) * 64; // O tile: rows om*16, cols on+
  const int rm = wid & 3, cb = wid >> 2;        // S tile: rows rm*16, cols cb*16

  const int kr = tid >> 3, kc = (tid & 7) * 16; // K stage slot
  const int vd = tid >> 1, vc = (tid & 1) * 16; // V stage slot

  auto stage = [&](int buf, int kv) {
    const _Float16* ksrc = K + base + (size_t)(kv + kr) * HD + kc;
    async_ld_b128(&Ks[buf][kr][kc],     ksrc);
    async_ld_b128(&Ks[buf][kr][kc + 8], ksrc + 8);
    const _Float16* vsrc = V + baseV + (size_t)vd * SEQ + kv + vc;
    async_ld_b128(&Vt[buf][vd][vc],     vsrc);
    async_ld_b128(&Vt[buf][vd][vc + 8], vsrc + 8);
  };

  // Q fragments straight from global into registers (A-fragment lane map)
  v16h qf[4];
  #pragma unroll
  for (int kk = 0; kk < 4; ++kk)
    qf[kk] = load_frag_a(Q + base + (size_t)(q0 + rm * 16) * HD + kk * 32, HD, lane);

  v8f o[4];
  #pragma unroll
  for (int j = 0; j < 4; ++j) { v8f z = {}; o[j] = z; }

  stage(0, 0);
  if (tid < 64) { m_l[tid] = -1e30f; l_l[tid] = 0.f; }

  const float sm = 0.08838834764831845f;   // 1/sqrt(128)

  for (int kv = 0; kv < SEQ; kv += 32) {
    int buf = (kv >> 5) & 1;
    if (kv + 32 < SEQ) { stage(buf ^ 1, kv + 32); WAIT_ASYNC(4); }
    else               { WAIT_ASYNC(0); }
    __syncthreads();

    { // S = Q @ K^T : one 16x16 tile per wave
      v8f sacc; { v8f z = {}; sacc = z; }
      #pragma unroll
      for (int kk = 0; kk < 4; ++kk) {
        v16h b = load_frag_b(&Ks[buf][cb * 16][kk * 32], 136, lane);
        sacc = wmma_f16(qf[kk], b, sacc);
      }
      #pragma unroll
      for (int r = 0; r < 8; ++r)
        Ss[rm * 16 + (lane >> 4) * 8 + r][cb * 16 + (lane & 15)] = sacc[r] * sm;
    }
    __syncthreads();

    if (tid < 64) { // online softmax, one thread per query row
      int row = tid;
      float mo = m_l[row], mx = mo;
      for (int j = 0; j < 32; ++j) mx = fmaxf(mx, Ss[row][j]);
      float sum = 0.f;
      for (int j = 0; j < 32; ++j) {
        float p = __expf(Ss[row][j] - mx);
        Ps[row][j] = (_Float16)p;
        sum += p;
      }
      float alpha = __expf(mo - mx);
      m_l[row]  = mx;
      l_l[row]  = l_l[row] * alpha + sum;
      al_l[row] = alpha;
    }
    __syncthreads();

    { // rescale O by alpha, then O += P @ V
      float alpha[8];
      #pragma unroll
      for (int r = 0; r < 8; ++r) alpha[r] = al_l[om * 16 + (lane >> 4) * 8 + r];
      #pragma unroll
      for (int j = 0; j < 4; ++j)
        #pragma unroll
        for (int r = 0; r < 8; ++r) o[j][r] *= alpha[r];

      v16h a = load_frag_a(&Ps[om * 16][0], 40, lane);
      #pragma unroll
      for (int j = 0; j < 4; ++j) {
        v16h b = load_frag_b(&Vt[buf][on + j * 16][0], 40, lane);
        o[j] = wmma_f16(a, b, o[j]);
      }
    }
    __syncthreads();
  }

  { // normalize by l and write [B,S,NH,HD]
    int b = bh >> 4, h = bh & 15;
    #pragma unroll
    for (int r = 0; r < 8; ++r) {
      int row = om * 16 + (lane >> 4) * 8 + r;
      float inv = 1.f / l_l[row];
      int s = q0 + row;
      size_t ob = (((size_t)b * SEQ + s) * NH + h) * HD;
      #pragma unroll
      for (int j = 0; j < 4; ++j)
        O[ob + on + j * 16 + (lane & 15)] = (_Float16)(o[j][r] * inv);
    }
  }
}

// ---------------- kernel 4: output projection  C = A @ Wo^T ------------------
__global__ __launch_bounds__(256) void oproj_gemm_kernel(
    const _Float16* __restrict__ A,    // [4096, 2048]
    const _Float16* __restrict__ Wo,   // [2048, 2048] row-major [n][k]
    float*          __restrict__ Cout) {
  __shared__ alignas(16) _Float16 As[2][64][40];
  __shared__ alignas(16) _Float16 Bt[2][128][40];

  const int tid = threadIdx.x, lane = tid & 31, wid = tid >> 5;
  const int wm = wid & 3, wn = wid >> 2;
  const int m0 = blockIdx.x * 64, n0 = blockIdx.y * 128;

  v8f acc[4];
  #pragma unroll
  for (int j = 0; j < 4; ++j) { v8f z = {}; acc[j] = z; }

  const int ar = tid >> 2, ac = (tid & 3) * 8;
  const int bn = tid & 127, bkh = tid >> 7;

  auto stage = [&](int buf, int kt) {
    async_ld_b128(&As[buf][ar][ac],
                  &A[(size_t)(m0 + ar) * HIDDEN + kt + ac]);
    const _Float16* src = &Wo[(size_t)(n0 + bn) * HIDDEN + kt + bkh * 16];
    async_ld_b128(&Bt[buf][bn][bkh * 16],     src);
    async_ld_b128(&Bt[buf][bn][bkh * 16 + 8], src + 8);
  };

  stage(0, 0);
  for (int kt = 0; kt < HIDDEN; kt += 32) {
    int buf = (kt >> 5) & 1;
    if (kt + 32 < HIDDEN) { stage(buf ^ 1, kt + 32); WAIT_ASYNC(3); }
    else                  { WAIT_ASYNC(0); }
    __syncthreads();

    v16h a = load_frag_a(&As[buf][wm * 16][0], 40, lane);
    #pragma unroll
    for (int j = 0; j < 4; ++j) {
      v16h b = load_frag_b(&Bt[buf][wn * 64 + j * 16][0], 40, lane);
      acc[j] = wmma_f16(a, b, acc[j]);
    }
    __syncthreads();
  }

  #pragma unroll
  for (int j = 0; j < 4; ++j)
    #pragma unroll
    for (int r = 0; r < 8; ++r) {
      int row = m0 + wm * 16 + (lane >> 4) * 8 + r;
      int col = n0 + wn * 64 + j * 16 + (lane & 15);
      Cout[(size_t)row * HIDDEN + col] = acc[j][r];
    }
}

// ---------------- host-side launch ------------------------------------------
extern "C" void kernel_launch(void* const* d_in, const int* in_sizes, int n_in,
                              void* d_out, int out_size, void* d_ws, size_t ws_size,
                              hipStream_t stream) {
  (void)in_sizes; (void)n_in; (void)out_size; (void)ws_size;

  const float* hs = (const float*)d_in[0];
  const float* Wq = (const float*)d_in[1];
  const float* Wk = (const float*)d_in[2];
  const float* Wv = (const float*)d_in[3];
  const float* Wo = (const float*)d_in[4];
  const float* qn = (const float*)d_in[5];
  const float* kn = (const float*)d_in[6];
  const float* vn = (const float*)d_in[7];
  const float* sc = (const float*)d_in[8];

  const size_t HS_N = (size_t)MTOT * HIDDEN;      // 8,388,608
  const size_t W_N  = (size_t)HIDDEN * HIDDEN;    // 4,194,304

  _Float16* ws   = (_Float16*)d_ws;
  _Float16* hs16 = ws;            ws += HS_N;
  _Float16* wqT  = ws;            ws += W_N;      // [n][k]
  _Float16* wkT  = ws;            ws += W_N;      // [n][k]
  _Float16* wvT  = ws;            ws += W_N;      // [n][k]
  _Float16* wo16 = ws;            ws += W_N;      // natural [n][k] already
  _Float16* q16  = ws;            ws += HS_N;     // [B,NH,S,HD]
  _Float16* k16  = ws;            ws += HS_N;     // [B,NH,S,HD]
  _Float16* v16t = ws;            ws += HS_N;     // [B,NH,HD,S]
  _Float16* ao16 = hs16;          // hs16 dead after QKV GEMMs — reuse

  cast_f32_to_f16<<<dim3((unsigned)(HS_N / 4 / 256)), 256, 0, stream>>>(hs, hs16, (int)HS_N);
  dim3 gt(HIDDEN / 64, HIDDEN / 64);
  cast_transpose_f32_to_f16<<<gt, 256, 0, stream>>>(Wq, wqT);
  cast_transpose_f32_to_f16<<<gt, 256, 0, stream>>>(Wk, wkT);
  cast_transpose_f32_to_f16<<<gt, 256, 0, stream>>>(Wv, wvT);
  cast_f32_to_f16<<<dim3((unsigned)(W_N / 4 / 256)), 256, 0, stream>>>(Wo, wo16, (int)W_N);

  dim3 g1(MTOT / 64, NH);
  qkv_gemm_kernel<<<g1, 256, 0, stream>>>(hs16, wqT, qn, sc, q16, 1);
  qkv_gemm_kernel<<<g1, 256, 0, stream>>>(hs16, wkT, kn, sc, k16, 1);
  qkv_gemm_kernel<<<g1, 256, 0, stream>>>(hs16, wvT, vn, sc, v16t, 0);

  dim3 g2(SEQ / 64, BATCH * NH);
  attn_kernel<<<g2, 256, 0, stream>>>(q16, k16, v16t, ao16);

  dim3 g3(MTOT / 64, HIDDEN / 128);
  oproj_gemm_kernel<<<g3, 256, 0, stream>>>(ao16, wo16, (float*)d_out);
}
